// RNNdiff_1425929142296
// MI455X (gfx1250) — matrix operand order repaired and checked by reading, not compiled, not run on hardware
//
#include <hip/hip_runtime.h>
#include <hip/hip_bf16.h>
#include <math.h>
#include <stdint.h>

typedef __attribute__((ext_vector_type(16))) _Float16 v16h;
typedef __attribute__((ext_vector_type(8)))  _Float16 v8h;
typedef __attribute__((ext_vector_type(8)))  float    v8f;

#define BSZ   256
#define VLEN  50
#define CDIM  20
#define DDIM  256
#define HDIM  256
#define G4H   1024
#define TDIFF 1000
#define NROWS (BSZ*VLEN)   /* 12800 */

// ---------------- WMMA fragment helpers (CDNA5 16x16x32 f16) ----------------
// A fragment: 16x32 f16 tile, row-major src, stride ld (halves).
// Per ISA: lanes 0-15 hold M=0..15 with K={0..7,16..23}; lanes 16-31 K={8..15,24..31}.
__device__ __forceinline__ v16h frag_a(const _Float16* p, int ld) {
  int lane = threadIdx.x & 31;
  int r = lane & 15, sel = lane >> 4;
  const _Float16* q = p + r * ld + sel * 8;
  v8h lo = *(const v8h*)q;          // K = 8*sel .. 8*sel+7
  v8h hi = *(const v8h*)(q + 16);   // K = 16+8*sel ..
  v16h out;
#pragma unroll
  for (int i = 0; i < 8; i++) { out[i] = lo[i]; out[i + 8] = hi[i]; }
  return out;
}
// B fragment: 32x16 f16 (K x N) with B stored transposed (N,K) row-major:
// lane holds col n = lane&15, K = {8*sel..8*sel+7, 16+8*sel..23+8*sel} (matches frag_a).
__device__ __forceinline__ v16h frag_b(const _Float16* bt, int ld) {
  int lane = threadIdx.x & 31;
  int n = lane & 15, sel = lane >> 4;
  const _Float16* q = bt + n * ld + sel * 8;
  v8h lo = *(const v8h*)q;
  v8h hi = *(const v8h*)(q + 16);
  v16h out;
#pragma unroll
  for (int i = 0; i < 8; i++) { out[i] = lo[i]; out[i + 8] = hi[i]; }
  return out;
}
__device__ __forceinline__ v8f wmma16(v16h a, v16h b, v8f c) {
  return __builtin_amdgcn_wmma_f32_16x16x32_f16(false, a, false, b, (short)0, c, false, false);
}
// Branch-free transcendentals: single v_exp_f32 / v_rcp_f32, correct saturation.
#define LOG2E 1.44269504088896340736f
__device__ __forceinline__ float sigm(float x) {
  return __builtin_amdgcn_rcpf(1.f + __builtin_amdgcn_exp2f(-x * LOG2E));
}
__device__ __forceinline__ float tanh_fast(float x) {
  return 1.f - 2.f * __builtin_amdgcn_rcpf(1.f + __builtin_amdgcn_exp2f(2.f * x * LOG2E));
}

// ---------------- register-blocked WMMA GEMM ----------------
// C(MxN,f32) = A(MxK,f16) @ Bt(NxK,f16)^T  [+ bias[col] (*rowscale[row])] [ReLU] [f16 mirror]
// Wave computes MT x NT 16x16 tiles: MT*NT independent accumulator chains,
// A frag reused NT times, B frag reused MT times.
// grid: (N/(NT*128), M/(MT*16)), 256 threads (8 waves).
template<int NT, int MT>
__global__ __launch_bounds__(256) void gemm_wmma(
    const _Float16* __restrict__ A, const _Float16* __restrict__ Bt,
    float* __restrict__ C, _Float16* __restrict__ Ch,
    const float* __restrict__ bias, const float* __restrict__ rowscale,
    int N, int K, int act)
{
  int wave = threadIdx.x >> 5, lane = threadIdx.x & 31;
  int col0 = (blockIdx.x * 8 + wave) * (16 * NT);
  int row0 = blockIdx.y * (16 * MT);
  const _Float16* Ap = A + (size_t)row0 * K;
  const _Float16* Bp = Bt + (size_t)col0 * K;
  v8f acc[MT][NT];
#pragma unroll
  for (int m = 0; m < MT; m++)
#pragma unroll
    for (int n = 0; n < NT; n++) { v8f z = {}; acc[m][n] = z; }
  for (int k0 = 0; k0 < K; k0 += 32) {
    v16h a[MT];
#pragma unroll
    for (int m = 0; m < MT; m++) {
      const _Float16* ap = Ap + (size_t)m * 16 * K + k0;
      a[m] = frag_a(ap, K);
      __builtin_prefetch(ap + 32, 0, 3);   // next k-tile of A
    }
#pragma unroll
    for (int n = 0; n < NT; n++) {
      v16h b = frag_b(Bp + (size_t)n * 16 * K + k0, K);
#pragma unroll
      for (int m = 0; m < MT; m++)
        acc[m][n] = wmma16(a[m], b, acc[m][n]);
    }
  }
  int r0 = lane & 15, sel = lane >> 4;
#pragma unroll
  for (int m = 0; m < MT; m++)
#pragma unroll
    for (int n = 0; n < NT; n++) {
      int col = col0 + n * 16 + r0;
      float bv = bias ? bias[col] : 0.f;
#pragma unroll
      for (int r = 0; r < 8; r++) {
        int row = row0 + m * 16 + r + 8 * sel;
        float bt = rowscale ? rowscale[row] * bv : bv;
        float v = acc[m][n][r] + bt;
        if (act) v = fmaxf(v, 0.f);
        size_t o = (size_t)row * N + col;
        C[o] = v;
        if (Ch) Ch[o] = (_Float16)v;
      }
    }
}

// ---------------- fused 50-step LSTM ----------------
// Block owns 16 batch rows (recurrence is batch-row independent -> all 50 steps in
// one launch).  Wave w computes gate tiles n = g*256 + 32w + 16h (g=0..3,h=0..1) so
// i/f/g/o for its 32 columns live in its own accumulators -> in-wave gating.
// Ping-pong LDS h buffer: read hS[cur], write hS[cur^1], ONE barrier per step.
// An opaque per-iteration *offset* keeps the 64 loop-invariant B-fragment loads
// inside the t-loop (no hoist-then-spill) while preserving the pointer's global
// address-space provenance (global_load, not flat_load).
__global__ __launch_bounds__(256) void lstm_wmma(
    const float* __restrict__ XW,      // (B*V, 1024) x@W_ih + (b_ih+b_hh), row=b*V+v
    const _Float16* __restrict__ WhhT, // (1024, 256) f16, (N,K)
    float* __restrict__ hOut,          // (B, V, 256) f32
    _Float16* __restrict__ hOutH)      // optional f16 mirror
{
  __shared__ _Float16 hS[2][16 * 256];
  int wave = threadIdx.x >> 5, lane = threadIdx.x & 31;
  int b0 = blockIdx.x * 16;
  int r0 = lane & 15, sel = lane >> 4;
  for (int i = threadIdx.x; i < 16 * 256; i += 256) hS[0][i] = (_Float16)0.f;
  float c[16];
#pragma unroll
  for (int i = 0; i < 16; i++) c[i] = 0.f;
  __syncthreads();
  int cur = 0;
  for (int t = 0; t < VLEN; t++) {
    // opaque zero offset, redefined every iteration -> B loads stay in-loop
    size_t zoff = 0;
    asm volatile("" : "+s"(zoff));
    const _Float16* wp = WhhT + zoff;
    // warm the XW gate slices for this step
    {
      int colG0 = wave * 32 + r0;
      const float* xw0 = XW + ((size_t)(b0 + 8 * sel) * VLEN + t) * G4H;
      const float* xw4 = XW + ((size_t)(b0 + 4 + 8 * sel) * VLEN + t) * G4H;
#pragma unroll
      for (int g = 0; g < 4; g++) {
        __builtin_prefetch(xw0 + g * 256 + colG0, 0, 3);
        __builtin_prefetch(xw4 + g * 256 + colG0, 0, 3);
      }
    }
    v8f acc[4][2];
#pragma unroll
    for (int g = 0; g < 4; g++)
#pragma unroll
      for (int h2 = 0; h2 < 2; h2++) { v8f z = {}; acc[g][h2] = z; }
    // rolled k-loop: 1 LDS A-frag + 8 global B-frags + 8 independent wmma per iter
#pragma unroll 1
    for (int k = 0; k < 8; k++) {
      v16h a = frag_a(hS[cur] + k * 32, 256);
#pragma unroll
      for (int g = 0; g < 4; g++)
#pragma unroll
        for (int h2 = 0; h2 < 2; h2++) {
          int n0 = g * 256 + wave * 32 + h2 * 16;
          v16h b = frag_b(wp + (size_t)n0 * 256 + k * 32, 256);
          acc[g][h2] = wmma16(a, b, acc[g][h2]);
        }
    }
    // gating: reads of hS[cur] are complete (frags consumed); write other buffer.
#pragma unroll
    for (int h2 = 0; h2 < 2; h2++) {
#pragma unroll
      for (int r = 0; r < 8; r++) {
        int rloc = r + 8 * sel;
        int brow = b0 + rloc;
        int colG = wave * 32 + h2 * 16 + r0;             // 0..255 within each gate
        const float* xw = XW + ((size_t)brow * VLEN + t) * G4H;
        float gi = acc[0][h2][r] + xw[colG];
        float gf = acc[1][h2][r] + xw[256 + colG];
        float gg = acc[2][h2][r] + xw[512 + colG];
        float go = acc[3][h2][r] + xw[768 + colG];
        float ci = sigm(gf) * c[h2 * 8 + r] + sigm(gi) * tanh_fast(gg);
        c[h2 * 8 + r] = ci;
        float h = sigm(go) * tanh_fast(ci);
        size_t o = ((size_t)brow * VLEN + t) * HDIM + colG;
        hOut[o] = h;
        if (hOutH) hOutH[o] = (_Float16)h;
        hS[cur ^ 1][rloc * 256 + colG] = (_Float16)h;
      }
    }
    __syncthreads();   // new h visible to all waves
    cur ^= 1;
  }
}

// ---------------- misc prep: cumprod(1-betas), b_ih+b_hh, t_norm rowscale ----------------
__global__ void prep_misc(const float* __restrict__ b_ih, const float* __restrict__ b_hh,
                          const int* __restrict__ tvec,
                          float* alphas, float* biasSum, float* rowscale)
{
  int tid = blockIdx.x * blockDim.x + threadIdx.x;
  if (tid == 0) {
    float a = 1.f;
    for (int t = 0; t < TDIFF; t++) {
      float beta = 0.0001f + (0.02f - 0.0001f) * ((float)t / (float)(TDIFF - 1));
      a *= (1.f - beta);
      alphas[t] = a;
    }
  }
  if (tid < G4H) biasSum[tid] = b_ih[tid] + b_hh[tid];
  if (tid < NROWS) rowscale[tid] = (float)tvec[tid / VLEN] / (float)TDIFF;
}

// src (K,N) f32 row-major -> dst (N,K) f16
__global__ void to_f16_T(const float* __restrict__ src, _Float16* __restrict__ dst, int K, int N)
{
  int i = blockIdx.x * 256 + threadIdx.x;
  if (i >= K * N) return;
  int k = i / N, n = i - k * N;
  dst[(size_t)n * K + k] = (_Float16)src[i];
}

// ---------------- embedding gather + time encoding ----------------
__global__ __launch_bounds__(256) void embed_kernel(
    const float* __restrict__ seq_time, const int* __restrict__ seqs,
    const float* __restrict__ embW, const float* __restrict__ w_time,
    const float* __restrict__ b_time, const float* __restrict__ w_updim,
    const float* __restrict__ b_updim,
    float* __restrict__ ve, _Float16* __restrict__ veH)
{
  int bv = blockIdx.x;                     // (b,v) pair
  __shared__ float tf[64];
  __shared__ int idx[CDIM];
  float ts = seq_time[bv] * (1.f / 180.f);
  if (threadIdx.x < 64) {
    float u = ts * w_time[threadIdx.x] + b_time[threadIdx.x];
    tf[threadIdx.x] = 1.f - tanh_fast(u * u);
  }
  if (threadIdx.x < CDIM) idx[threadIdx.x] = seqs[bv * CDIM + threadIdx.x];
  __syncthreads();
  int cc = threadIdx.x;
  float acc = b_updim[cc];
#pragma unroll 8
  for (int j = 0; j < 64; j++) acc += tf[j] * w_updim[j * DDIM + cc];
#pragma unroll 4
  for (int i = 0; i < CDIM; i++) acc += fmaxf(embW[(size_t)idx[i] * DDIM + cc], 0.f);
  ve[(size_t)bv * DDIM + cc] = acc;
  veH[(size_t)bv * DDIM + cc] = (_Float16)acc;
}

// barE[:,0,:] = ve[:,0,:]
__global__ void bar0_kernel(const float* __restrict__ ve, float* __restrict__ barE)
{
  int i = blockIdx.x * 256 + threadIdx.x;  // B*256
  int b = i >> 8, cc = i & 255;
  barE[(size_t)b * VLEN * DDIM + cc] = ve[(size_t)b * VLEN * DDIM + cc];
}

// ---------------- attention + bar_rest (one wave per (b,v), v<49) ----------------
__global__ __launch_bounds__(256) void attn_kernel(
    const float* __restrict__ ve, const float* __restrict__ wh,
    const float* __restrict__ W1, const float* __restrict__ b1,
    const float* __restrict__ W2, const float* __restrict__ b2,
    float* __restrict__ barE)
{
  int gw = (blockIdx.x * 256 + threadIdx.x) >> 5;
  int lane = threadIdx.x & 31;
  if (gw >= BSZ * (VLEN - 1)) return;
  int b = gw / (VLEN - 1), v = gw - b * (VLEN - 1);
  const float* e0 = ve + (size_t)b * VLEN * DDIM;
  const float* whr = wh + ((size_t)b * VLEN + v) * DDIM;
  float y0 = b1[lane], y1 = b1[lane + 32];
  for (int k = 0; k < 256; k++) {
    float xe = e0[k], xw = whr[k];
    y0 += xe * W1[k * 64 + lane] + xw * W1[(256 + k) * 64 + lane];
    y1 += xe * W1[k * 64 + lane + 32] + xw * W1[(256 + k) * 64 + lane + 32];
  }
  y0 = tanh_fast(y0); y1 = tanh_fast(y1);
  float a0 = y0 * W2[lane * 2 + 0] + y1 * W2[(lane + 32) * 2 + 0];
  float a1 = y0 * W2[lane * 2 + 1] + y1 * W2[(lane + 32) * 2 + 1];
  for (int off = 16; off; off >>= 1) {
    a0 += __shfl_down(a0, off, 32);
    a1 += __shfl_down(a1, off, 32);
  }
  a0 = __shfl(a0, 0, 32) + b2[0];
  a1 = __shfl(a1, 0, 32) + b2[1];
  float* dst = barE + ((size_t)b * VLEN + v + 1) * DDIM;
  for (int cc = lane; cc < 256; cc += 32) dst[cc] = e0[cc] * a0 + whr[cc] * a1;
}

// e_noisy (f16 for GEMM) + pass-through noise output
__global__ void noise_kernel(const float* __restrict__ barE, const float* __restrict__ nz,
                             const float* __restrict__ alphas, const int* __restrict__ tvec,
                             _Float16* __restrict__ eH, float* __restrict__ outNz)
{
  int i = blockIdx.x * 256 + threadIdx.x;     // < NROWS*256
  int b = i / (VLEN * DDIM);
  float a = alphas[tvec[b]];
  float sa = sqrtf(a), sn = sqrtf(1.f - a);
  float n = nz[i];
  eH[i] = (_Float16)(barE[i] * sa + n * sn);
  outNz[i] = n;
}

// gen = bar_e_k + noise - predicted_noise  (f16 for X@W_ih GEMM)
__global__ void gen_kernel(const float* __restrict__ barE, const float* __restrict__ nz,
                           const float* __restrict__ pred, _Float16* __restrict__ genH)
{
  int i = blockIdx.x * 256 + threadIdx.x;
  genH[i] = (_Float16)(barE[i] + nz[i] - pred[i]);
}

// gather hidden[:, -1] and hidden_gen[:, -1] into a 512x256 f16 matrix
__global__ void clsa_kernel(const float* __restrict__ hid, const float* __restrict__ hidg,
                            _Float16* __restrict__ A)
{
  int i = blockIdx.x * 256 + threadIdx.x;     // 512*256
  int row = i >> 8, cc = i & 255;
  float v = (row < BSZ) ? hid[((size_t)row * VLEN + (VLEN - 1)) * DDIM + cc]
                        : hidg[((size_t)(row - BSZ) * VLEN + (VLEN - 1)) * DDIM + cc];
  A[i] = (_Float16)v;
}

// final classifier layer: (512,512) @ (512,2) -> two d_out slots
__global__ void cls3_kernel(const float* __restrict__ h2, const float* __restrict__ Wc3,
                            const float* __restrict__ bc3,
                            float* __restrict__ out_h, float* __restrict__ out_g)
{
  int gw = (blockIdx.x * 256 + threadIdx.x) >> 5;
  int lane = threadIdx.x & 31;
  if (gw >= 512) return;
  const float* row = h2 + (size_t)gw * 512;
  float s0 = 0.f, s1 = 0.f;
  for (int k = lane; k < 512; k += 32) {
    float x = row[k];
    s0 += x * Wc3[k * 2 + 0];
    s1 += x * Wc3[k * 2 + 1];
  }
  for (int off = 16; off; off >>= 1) {
    s0 += __shfl_down(s0, off, 32);
    s1 += __shfl_down(s1, off, 32);
  }
  if (lane == 0) {
    float o0 = s0 + bc3[0], o1 = s1 + bc3[1];
    if (gw < BSZ) { out_h[gw * 2] = o0; out_h[gw * 2 + 1] = o1; }
    else          { out_g[(gw - BSZ) * 2] = o0; out_g[(gw - BSZ) * 2 + 1] = o1; }
  }
}

// ------------------------------------------------------------------
extern "C" void kernel_launch(void* const* d_in, const int* in_sizes, int n_in,
                              void* d_out, int out_size, void* d_ws, size_t ws_size,
                              hipStream_t stream)
{
  (void)in_sizes; (void)n_in; (void)out_size; (void)ws_size;
  const float* seq_time = (const float*)d_in[0];
  const float* noise    = (const float*)d_in[1];
  const float* embW     = (const float*)d_in[2];
  const float* w_time   = (const float*)d_in[3];
  const float* b_time   = (const float*)d_in[4];
  const float* w_updim  = (const float*)d_in[5];
  const float* b_updim  = (const float*)d_in[6];
  const float* W_ih     = (const float*)d_in[7];
  const float* W_hh     = (const float*)d_in[8];
  const float* b_ih     = (const float*)d_in[9];
  const float* b_hh     = (const float*)d_in[10];
  const float* W_hk     = (const float*)d_in[11];
  const float* b_hk     = (const float*)d_in[12];
  const float* W1       = (const float*)d_in[13];
  const float* b1       = (const float*)d_in[14];
  const float* W2       = (const float*)d_in[15];
  const float* b2       = (const float*)d_in[16];
  const float* Wc1      = (const float*)d_in[17];
  const float* bc1      = (const float*)d_in[18];
  const float* Wc2      = (const float*)d_in[19];
  const float* bc2      = (const float*)d_in[20];
  const float* Wc3      = (const float*)d_in[21];
  const float* bc3      = (const float*)d_in[22];
  const float* W_diff   = (const float*)d_in[23];
  const float* b_diff   = (const float*)d_in[24];
  const int*   seqs     = (const int*)d_in[25];
  const int*   tvec     = (const int*)d_in[26];

  float* out_hidden = (float*)d_out;
  float* out_hidgen = out_hidden + (size_t)NROWS * HDIM;
  float* out_cls_h  = out_hidgen + (size_t)NROWS * HDIM;
  float* out_cls_g  = out_cls_h + BSZ * 2;
  float* out_noise  = out_cls_g + BSZ * 2;
  float* out_pred   = out_noise + (size_t)NROWS * DDIM;

  char* ws = (char*)d_ws;
  size_t off = 0;
  auto alloc = [&](size_t bytes) -> char* {
    char* p = ws + off;
    off += (bytes + 255) & ~(size_t)255;
    return p;
  };
  float*    alphas   = (float*)alloc(TDIFF * 4);
  float*    biasSum  = (float*)alloc(G4H * 4);
  float*    rowscale = (float*)alloc((size_t)NROWS * 4);
  _Float16* W_ihT    = (_Float16*)alloc((size_t)G4H * DDIM * 2);
  _Float16* W_hhT    = (_Float16*)alloc((size_t)G4H * HDIM * 2);
  _Float16* W_hkT    = (_Float16*)alloc((size_t)DDIM * HDIM * 2);
  _Float16* Wc1T     = (_Float16*)alloc((size_t)G4H * HDIM * 2);
  _Float16* Wc2T     = (_Float16*)alloc((size_t)512 * G4H * 2);
  _Float16* W_diffT  = (_Float16*)alloc((size_t)DDIM * DDIM * 2);
  float*    ve       = (float*)alloc((size_t)NROWS * DDIM * 4);
  _Float16* veH      = (_Float16*)alloc((size_t)NROWS * DDIM * 2);
  float*    XW       = (float*)alloc((size_t)NROWS * G4H * 4);
  _Float16* hiddenH  = (_Float16*)alloc((size_t)NROWS * HDIM * 2);
  float*    wh       = (float*)alloc((size_t)NROWS * DDIM * 4);
  float*    barE     = (float*)alloc((size_t)NROWS * DDIM * 4);
  _Float16* eH       = (_Float16*)alloc((size_t)NROWS * DDIM * 2);
  _Float16* genH     = (_Float16*)alloc((size_t)NROWS * DDIM * 2);
  _Float16* clsAH    = (_Float16*)alloc((size_t)512 * DDIM * 2);
  float*    h1       = (float*)alloc((size_t)512 * G4H * 4);
  _Float16* h1H      = (_Float16*)alloc((size_t)512 * G4H * 2);
  float*    h2       = (float*)alloc((size_t)512 * 512 * 4);

  // one-time prep (replayed each call; cheap)
  prep_misc<<<50, 256, 0, stream>>>(b_ih, b_hh, tvec, alphas, biasSum, rowscale);
  to_f16_T<<<(DDIM * G4H) / 256, 256, 0, stream>>>(W_ih,  W_ihT,  DDIM, G4H);
  to_f16_T<<<(HDIM * G4H) / 256, 256, 0, stream>>>(W_hh,  W_hhT,  HDIM, G4H);
  to_f16_T<<<(HDIM * DDIM) / 256, 256, 0, stream>>>(W_hk, W_hkT,  HDIM, DDIM);
  to_f16_T<<<(HDIM * G4H) / 256, 256, 0, stream>>>(Wc1,   Wc1T,   HDIM, G4H);
  to_f16_T<<<(G4H * 512) / 256, 256, 0, stream>>>(Wc2,    Wc2T,   G4H, 512);
  to_f16_T<<<(DDIM * DDIM) / 256, 256, 0, stream>>>(W_diff, W_diffT, DDIM, DDIM);

  // ve = relu(emb).sum + time_encoding
  embed_kernel<<<NROWS, 256, 0, stream>>>(seq_time, seqs, embW, w_time, b_time,
                                          w_updim, b_updim, ve, veH);
  // XW = ve @ W_ih + (b_ih+b_hh)   (M=12800,K=256,N=1024): 8 waves x (2x8) tiles
  gemm_wmma<8, 2><<<dim3(1, NROWS / 32), 256, 0, stream>>>(
      veH, W_ihT, XW, nullptr, biasSum, nullptr, G4H, DDIM, 0);
  // LSTM #1 (all 50 steps, one launch)
  lstm_wmma<<<16, 256, 0, stream>>>(XW, W_hhT, out_hidden, hiddenH);
  // wh = hidden @ W_hk + b_hk  (computed for all v; attn uses v<49)
  gemm_wmma<2, 2><<<dim3(1, NROWS / 32), 256, 0, stream>>>(
      hiddenH, W_hkT, wh, nullptr, b_hk, nullptr, DDIM, HDIM, 0);
  bar0_kernel<<<256, 256, 0, stream>>>(ve, barE);
  attn_kernel<<<(BSZ * (VLEN - 1)) / 8, 256, 0, stream>>>(ve, wh, W1, b1, W2, b2, barE);
  // e_noisy (f16) + noise passthrough output
  noise_kernel<<<NROWS, 256, 0, stream>>>(barE, noise, alphas, tvec, eH, out_noise);
  // predicted_noise = e_noisy @ W_diff + t_norm * b_diff
  gemm_wmma<2, 2><<<dim3(1, NROWS / 32), 256, 0, stream>>>(
      eH, W_diffT, out_pred, nullptr, b_diff, rowscale, DDIM, DDIM, 0);
  gen_kernel<<<NROWS, 256, 0, stream>>>(barE, noise, out_pred, genH);
  // XW = gen @ W_ih + bias  (reuse buffer)
  gemm_wmma<8, 2><<<dim3(1, NROWS / 32), 256, 0, stream>>>(
      genH, W_ihT, XW, nullptr, biasSum, nullptr, G4H, DDIM, 0);
  // LSTM #2
  lstm_wmma<<<16, 256, 0, stream>>>(XW, W_hhT, out_hidgen, nullptr);
  // classifier heads (both branches batched: 512 rows)
  clsa_kernel<<<512, 256, 0, stream>>>(out_hidden, out_hidgen, clsAH);
  gemm_wmma<8, 2><<<dim3(1, 512 / 32), 256, 0, stream>>>(
      clsAH, Wc1T, h1, h1H, bc1, nullptr, G4H, HDIM, 1);
  gemm_wmma<4, 2><<<dim3(1, 512 / 32), 256, 0, stream>>>(
      h1H, Wc2T, h2, nullptr, bc2, nullptr, 512, G4H, 1);
  cls3_kernel<<<64, 256, 0, stream>>>(h2, Wc3, bc3, out_cls_h, out_cls_g);
}
